// CELossTotalEval_33629593928291
// MI455X (gfx1250) — compile-verified, matrix operand order checked by reading, not run on hardware
//
#include <hip/hip_runtime.h>
#include <hip/hip_bf16.h>
#include <math.h>

// Problem constants (match reference)
#define N_    16
#define T_    256
#define V_    16384
#define NGRAM_ 4
#define ROWS_ (N_ * T_)        // 4096 rows per tensor
#define IGNORE_ (-1)

typedef float v4f __attribute__((ext_vector_type(4)));
typedef int   v4i __attribute__((ext_vector_type(4)));
typedef int   v8i __attribute__((ext_vector_type(8)));

#ifndef __has_builtin
#define __has_builtin(x) 0
#endif

#if __has_builtin(__builtin_amdgcn_global_load_async_to_lds_b128)
#define HAVE_ASYNC_LOAD 1
#else
#define HAVE_ASYNC_LOAD 0
#endif
#if __has_builtin(__builtin_amdgcn_s_wait_asynccnt)
#define HAVE_ASYNC_WAIT 1
#else
#define HAVE_ASYNC_WAIT 0
#endif

#if HAVE_ASYNC_LOAD && HAVE_ASYNC_WAIT
#define USE_ASYNC 1
#else
#define USE_ASYNC 0
#endif

#if USE_ASYNC
__device__ __forceinline__ void async_copy16(const void* g, void* l) {
  // global -> LDS 16B async copy, tracked by ASYNCcnt (param type is int4*).
  __builtin_amdgcn_global_load_async_to_lds_b128((v4i*)g, (v4i*)l, 0, 0);
}
template <int NN>
__device__ __forceinline__ void wait_asynccnt() {
  __builtin_amdgcn_s_wait_asynccnt(NN);
}
#else
template <int NN>
__device__ __forceinline__ void wait_asynccnt() {}
#endif

// accumulate a float4 (elements e0..e0+3) into sum / max / argmax (first index wins ties)
#define ACCUM4(vv, ee)                                          \
  do {                                                          \
    s += (vv).x + (vv).y + (vv).z + (vv).w;                     \
    if ((vv).x > m) { m = (vv).x; mi = (ee) + 0; }              \
    if ((vv).y > m) { m = (vv).y; mi = (ee) + 1; }              \
    if ((vv).z > m) { m = (vv).z; mi = (ee) + 2; }              \
    if ((vv).w > m) { m = (vv).w; mi = (ee) + 3; }              \
  } while (0)

// ---------------------------------------------------------------------------
// Kernel 1: per-row reduction. grid = (4096 rows, 3 tensors), 256 threads.
// Streams one 64KB row via async global->LDS 4-deep ring (or NT b128 loads).
// Outputs per row: rowsum, p[target]; for tensor 0 also max + argmax.
// ---------------------------------------------------------------------------
__global__ __launch_bounds__(256) void rowstat_kernel(
    const float* __restrict__ out0, const float* __restrict__ out1,
    const float* __restrict__ out2, const int* __restrict__ tgt0,
    const int* __restrict__ tgt1, float* __restrict__ sumArr,
    float* __restrict__ ptgtArr, float* __restrict__ maxArr,
    int* __restrict__ amaxArr) {
  const int tid = threadIdx.x;
  const int row = blockIdx.x;    // 0..4095  (= n*T + t)
  const int which = blockIdx.y;  // 0..2
  const float* src = (which == 0) ? out0 : ((which == 1) ? out1 : out2);
  const size_t base = (size_t)row * V_;

  float s = 0.0f;
  float m = -3.402823466e38f;
  int mi = 0;

#if USE_ASYNC
  // 16 chunks of 1024 floats (4KB); each lane moves its own 16B per chunk,
  // so per-wave s_wait_asynccnt is sufficient (no cross-lane sharing).
  __shared__ v4f stage[4][256];  // 16 KB ring
  const char* g = (const char*)(src + base);
  async_copy16(g + (size_t)0 * 4096 + tid * 16, &stage[0][tid]);
  async_copy16(g + (size_t)1 * 4096 + tid * 16, &stage[1][tid]);
  async_copy16(g + (size_t)2 * 4096 + tid * 16, &stage[2][tid]);
  async_copy16(g + (size_t)3 * 4096 + tid * 16, &stage[3][tid]);
#pragma unroll 4
  for (int c = 0; c < 12; ++c) {
    wait_asynccnt<3>();  // issued = c+4, wait<=3  =>  chunk c complete (in-order)
    v4f v = stage[c & 3][tid];
    int e0 = c * 1024 + tid * 4;
    ACCUM4(v, e0);
    async_copy16(g + (size_t)(c + 4) * 4096 + tid * 16, &stage[c & 3][tid]);
  }
  {
    wait_asynccnt<3>(); v4f v = stage[12 & 3][tid]; ACCUM4(v, 12 * 1024 + tid * 4);
  }
  {
    wait_asynccnt<2>(); v4f v = stage[13 & 3][tid]; ACCUM4(v, 13 * 1024 + tid * 4);
  }
  {
    wait_asynccnt<1>(); v4f v = stage[14 & 3][tid]; ACCUM4(v, 14 * 1024 + tid * 4);
  }
  {
    wait_asynccnt<0>(); v4f v = stage[15 & 3][tid]; ACCUM4(v, 15 * 1024 + tid * 4);
  }
#else
  // Fallback: direct non-temporal b128 streaming loads (read-once data,
  // keep it out of the 192MB L2).
  const v4f* g4 = (const v4f*)(src + base);
  for (int i = tid; i < V_ / 4; i += 256) {
    v4f v = __builtin_nontemporal_load(g4 + i);
    int e0 = i * 4;
    ACCUM4(v, e0);
  }
#endif

  // block tree-reduce: sum, (max, argmax) with first-index tie-break
  __shared__ float ssum[256];
  __shared__ float smax[256];
  __shared__ int sidx[256];
  ssum[tid] = s;
  smax[tid] = m;
  sidx[tid] = mi;
  __syncthreads();
  for (int w = 128; w > 0; w >>= 1) {
    if (tid < w) {
      ssum[tid] += ssum[tid + w];
      float om = smax[tid + w];
      int oi = sidx[tid + w];
      if (om > smax[tid] || (om == smax[tid] && oi < sidx[tid])) {
        smax[tid] = om;
        sidx[tid] = oi;
      }
    }
    __syncthreads();
  }

  if (tid == 0) {
    sumArr[which * ROWS_ + row] = ssum[0];
    if (which == 0) {
      maxArr[row] = smax[0];
      amaxArr[row] = sidx[0];
    }
    // p[target] gather (one scalar load per row)
    int tIn = row % T_;
    int tg;
    if (which == 0) {
      tg = (tIn < T_ - 1) ? tgt0[row + 1] : IGNORE_;  // CE-shift target
    } else {
      tg = tgt1[row];
    }
    float pt = (tg >= 0 && tg < V_) ? src[base + (size_t)tg] : 1.0f;
    ptgtArr[which * ROWS_ + row] = pt;
  }
}

// ---------------------------------------------------------------------------
// Kernel 2: single block (8 wave32s).
// 4-gram repeat detection via IU8 WMMA Gram matrix:
//   each 4-gram of 14-bit tokens -> 8 u8 bytes (lo/hi per token), zero-padded
//   to K=64. ngram j == ngram j'  <=>  sa[j] + sa[j'] - 2*dot(j,j') == 0
//   (exact integer arithmetic, max dot = 8*255^2).
// Then UL masked sum + three CE means.
// ---------------------------------------------------------------------------
__global__ __launch_bounds__(256) void finalize_kernel(
    const float* __restrict__ sumArr, const float* __restrict__ ptgtArr,
    const float* __restrict__ maxArr, const int* __restrict__ amaxArr,
    const int* __restrict__ tgt0, const int* __restrict__ tgt1,
    float* __restrict__ out) {
  const int tid = threadIdx.x;  // 256 == T_
  const int wid = tid >> 5;     // wave id 0..7
  const int lane = tid & 31;
  __shared__ int predRow[T_];
  __shared__ unsigned int ngB0[T_];   // ngram bytes K0..K3
  __shared__ unsigned int ngB1[T_];   // ngram bytes K4..K7
  __shared__ int saS[T_];             // sum of squared bytes
  __shared__ unsigned char repS[T_];
  __shared__ double red[256];

  // ---- unlikelihood loss -------------------------------------------------
  double ul = 0.0;
  const int J = T_ - NGRAM_;  // 252
  const int NTILE = (16 * 17) / 2;  // 136 lower-triangle 16x16 tiles (exactly 17/wave)
  for (int n = 0; n < N_; ++n) {
    predRow[tid] = amaxArr[n * T_ + tid];
    repS[tid] = 0;
    __syncthreads();

    // build byte-expanded 4-grams (zero for padded j >= J)
    unsigned int b0 = 0u, b1 = 0u;
    int sa = 0;
    if (tid < J) {
      int t0 = predRow[tid], t1 = predRow[tid + 1];
      int t2 = predRow[tid + 2], t3 = predRow[tid + 3];
      unsigned int l0 = t0 & 255u, h0 = ((unsigned)t0 >> 8) & 255u;
      unsigned int l1 = t1 & 255u, h1 = ((unsigned)t1 >> 8) & 255u;
      unsigned int l2 = t2 & 255u, h2 = ((unsigned)t2 >> 8) & 255u;
      unsigned int l3 = t3 & 255u, h3 = ((unsigned)t3 >> 8) & 255u;
      b0 = l0 | (h0 << 8) | (l1 << 16) | (h1 << 24);
      b1 = l2 | (h2 << 8) | (l3 << 16) | (h3 << 24);
      sa = (int)(l0 * l0 + h0 * h0 + l1 * l1 + h1 * h1 +
                 l2 * l2 + h2 * h2 + l3 * l3 + h3 * h3);
    }
    ngB0[tid] = b0;
    ngB1[tid] = b1;
    saS[tid] = sa;
    __syncthreads();

    // WMMA pass over lower-triangle tiles: tile (i, jj<=i) covers
    // rows j = i*16+m, cols j' = jj*16+nn.
    for (int tix = wid; tix < NTILE; tix += 8) {
      int i = 0, accum = 0;
      while (tix >= accum + i + 1) { accum += i + 1; ++i; }
      int jj = tix - accum;

      // A: lanes 0-15 carry M=0..15 with K0-3 in VGPR0, K4-7 in VGPR1
      //    (lanes 16-31 = K8..15 -> zero padding). B: same packing per column.
      int ja = i * 16 + (lane & 15);
      int jb = jj * 16 + (lane & 15);
      int a0 = 0, a1 = 0, bb0 = 0, bb1 = 0;
      if (lane < 16) {
        a0 = (int)ngB0[ja]; a1 = (int)ngB1[ja];
        bb0 = (int)ngB0[jb]; bb1 = (int)ngB1[jb];
      }
      v8i A = {a0, a1, 0, 0, 0, 0, 0, 0};
      v8i B = {bb0, bb1, 0, 0, 0, 0, 0, 0};
      v8i C = {0, 0, 0, 0, 0, 0, 0, 0};
      // v_wmma_i32_16x16x64_iu8: D = A(16x64 u8) * B(64x16 u8) + C
      v8i D = __builtin_amdgcn_wmma_i32_16x16x64_iu8(false, A, false, B, C,
                                                     false, false);
      // D layout: VGPR r -> M = r (lanes 0-15) / r+8 (lanes 16-31), N = lane&15
      int j2 = jj * 16 + (lane & 15);
      int mbase = i * 16 + ((lane < 16) ? 0 : 8);
      int sb = saS[j2];
#pragma unroll
      for (int r = 0; r < 8; ++r) {
        int j = mbase + r;
        if (j < J && j2 < j && (saS[j] + sb == 2 * D[r])) repS[j] = 1;
      }
    }
    __syncthreads();

    // mask[t] = OR of rep[t-3..t] (valid j range), then UL term
    bool msk = false;
#pragma unroll
    for (int k = 0; k < NGRAM_; ++k) {
      int j = tid - k;
      if (j >= 0 && j < J && repS[j]) msk = true;
    }
    if (msk) {
      float g = maxArr[n * T_ + tid];  // probs at argmax
      float om = fmaxf(1.0f - expf(g), 1e-20f);
      ul += -(double)logf(om);
    }
    __syncthreads();
  }

  // ---- cross entropies ---------------------------------------------------
  // nll = log(sum p) - log(p[target]); mean over valid targets
  double ce0s = 0.0, ce1s = 0.0, ce2s = 0.0;
  int c0 = 0, c1 = 0;
  for (int r = tid; r < ROWS_; r += 256) {
    int t = r % T_;
    if (t < T_ - 1) {
      int tg = tgt0[r + 1];
      if (tg != IGNORE_) {
        ce0s += log((double)sumArr[0 * ROWS_ + r]) -
                log((double)ptgtArr[0 * ROWS_ + r]);
        ++c0;
      }
    }
    int tg1 = tgt1[r];
    if (tg1 != IGNORE_) {
      ce1s += log((double)sumArr[1 * ROWS_ + r]) -
              log((double)ptgtArr[1 * ROWS_ + r]);
      ce2s += log((double)sumArr[2 * ROWS_ + r]) -
              log((double)ptgtArr[2 * ROWS_ + r]);
      ++c1;
    }
  }

  // ---- block reduce 6 scalars -------------------------------------------
  double acc[6] = {ce0s, (double)c0, ce1s, ce2s, (double)c1, ul};
  double tot[6];
  for (int q = 0; q < 6; ++q) {
    __syncthreads();
    red[tid] = acc[q];
    __syncthreads();
    for (int w = 128; w > 0; w >>= 1) {
      if (tid < w) red[tid] += red[tid + w];
      __syncthreads();
    }
    if (tid == 0) tot[q] = red[0];
  }
  if (tid == 0) {
    double r = tot[0] / tot[1]   // ce_shift(out0, tgt0)
             + tot[5]            // ul(out0)
             + tot[2] / tot[4]   // ce(out1, tgt1)
             + tot[3] / tot[4];  // ce(out2, tgt1)
    out[0] = (float)r;
  }
}

// ---------------------------------------------------------------------------
extern "C" void kernel_launch(void* const* d_in, const int* in_sizes, int n_in,
                              void* d_out, int out_size, void* d_ws,
                              size_t ws_size, hipStream_t stream) {
  (void)in_sizes; (void)n_in; (void)out_size; (void)ws_size;
  const float* out0 = (const float*)d_in[0];
  const float* out1 = (const float*)d_in[1];
  const float* out2 = (const float*)d_in[2];
  const int* tgt0 = (const int*)d_in[3];
  const int* tgt1 = (const int*)d_in[4];
  float* out = (float*)d_out;

  // workspace layout (needs 8*4096*4 = 128 KB)
  float* ws = (float*)d_ws;
  float* sumArr = ws;                 // 3 * 4096 f32
  float* ptgtArr = ws + 3 * ROWS_;    // 3 * 4096 f32
  float* maxArr = ws + 6 * ROWS_;     // 4096 f32
  int* amaxArr = (int*)(ws + 7 * ROWS_);  // 4096 i32

  dim3 gridA(ROWS_, 3, 1);
  rowstat_kernel<<<gridA, 256, 0, stream>>>(out0, out1, out2, tgt0, tgt1,
                                            sumArr, ptgtArr, maxArr, amaxArr);
  finalize_kernel<<<1, 256, 0, stream>>>(sumArr, ptgtArr, maxArr, amaxArr,
                                         tgt0, tgt1, out);
}
